// MultiHeadAttention_47966194761863
// MI455X (gfx1250) — matrix operand order repaired and checked by reading, not compiled, not run on hardware
//
#include <hip/hip_runtime.h>

// ---------------------------------------------------------------------------
// MHA forward for MI455X (gfx1250): bf16 WMMA everywhere, fp32 accumulate.
// Outputs (concatenated in d_out): final [B,T,C] fp32, weights [B,H,T,T] fp32.
// GEMMs use async-to-LDS double-buffered staging (ASYNCcnt) when available.
// ---------------------------------------------------------------------------

typedef __attribute__((ext_vector_type(16))) __bf16 v16bf;
typedef __attribute__((ext_vector_type(8)))  __bf16 v8bf;
typedef __attribute__((ext_vector_type(4)))  __bf16 v4bf;
typedef __attribute__((ext_vector_type(8)))  float  v8f;
typedef int v4i_vs __attribute__((vector_size(16)));   // matches builtin param type

#define B_SZ 2
#define T_SZ 2048
#define C_SZ 768
#define H_SZ 12
#define DH   64
#define M_SZ (B_SZ * T_SZ)   // 4096 rows of x / q / k / v / attn_out

#define AS_GLOBAL __attribute__((address_space(1)))
#define AS_LOCAL  __attribute__((address_space(3)))

#if __has_builtin(__builtin_amdgcn_global_load_async_to_lds_b128) && \
    __has_builtin(__builtin_amdgcn_s_wait_asynccnt)
#define USE_ASYNC_LDS 1
#else
#define USE_ASYNC_LDS 0
#endif

__device__ __forceinline__ v8bf ld8(const __bf16* p) { return *(const v8bf*)p; }

#if USE_ASYNC_LDS
// one 16-byte async global->LDS transfer per lane (tracked by ASYNCcnt)
__device__ __forceinline__ void async_ld_b128(const void* gsrc, void* ldst) {
  __builtin_amdgcn_global_load_async_to_lds_b128(
      (AS_GLOBAL v4i_vs*)(unsigned long long)gsrc,
      (AS_LOCAL  v4i_vs*)(unsigned int)(unsigned long long)ldst, 0, 0);
}
#endif

// ---------------------------------------------------------------------------
// fp32 -> bf16 conversion (vectorized x4)
// ---------------------------------------------------------------------------
__global__ void cvt_f32_bf16(const float* __restrict__ in, __bf16* __restrict__ out, int n4) {
  int i = blockIdx.x * blockDim.x + threadIdx.x;
  if (i < n4) {
    float4 v = ((const float4*)in)[i];
    v4bf o;
    o[0] = (__bf16)v.x; o[1] = (__bf16)v.y; o[2] = (__bf16)v.z; o[3] = (__bf16)v.w;
    ((v4bf*)out)[i] = o;
  }
}

// ---------------------------------------------------------------------------
// Generic out = X * W^T GEMM. X:[M][K] bf16 row-major, W:[N][K] bf16 row-major
// (so B[k][n] = W[n][k], making B-fragments two contiguous 16B loads per lane).
// mode 0: bf16 row-major [M][N]   (Q, K projections)
// mode 1: bf16 transposed [N][M]  (V projection -> Vt, contiguous b128 stores)
// mode 2: fp32 row-major [M][N]   (final O projection into d_out)
// Block = 256 thr (8 waves): block tile 64(M) x 64(N); wave tile 16 x 32.
// A/W 64x32 k-slices are double-buffered in LDS via async-to-LDS copies.
// ---------------------------------------------------------------------------
__global__ __launch_bounds__(256)
void gemm_xwT_wmma(const __bf16* __restrict__ A, const __bf16* __restrict__ W,
                   void* __restrict__ outp, int M, int N, int K, int mode, float scale) {
  const int tid  = threadIdx.x;
  const int wave = tid >> 5, lane = tid & 31;
  const int lm = lane & 15, lh = lane >> 4;
  const int mw = wave & 3,  nw = wave >> 2;
  const int m0 = blockIdx.x * 64 + mw * 16;
  const int n0 = blockIdx.y * 64 + nw * 32;

  v8f acc0 = {}; v8f acc1 = {};

#if USE_ASYNC_LDS
  __shared__ __bf16 As[2][64 * 32];   // 4KB per buffer
  __shared__ __bf16 Bs[2][64 * 32];

  const int srow = tid >> 2;          // 0..63 : tile row staged by this thread
  const int scol = (tid & 3) * 8;     // 0/8/16/24 : 16B chunk within the row
  const __bf16* Ag = A + (size_t)(blockIdx.x * 64 + srow) * K + scol;
  const __bf16* Bg = W + (size_t)(blockIdx.y * 64 + srow) * K + scol;

  // preload k0 = 0 into buffer 0
  async_ld_b128(Ag, &As[0][srow * 32 + scol]);
  async_ld_b128(Bg, &Bs[0][srow * 32 + scol]);

  int buf = 0;
  for (int k0 = 0; k0 < K; k0 += 32, buf ^= 1) {
    __builtin_amdgcn_s_wait_asynccnt(0);
    __syncthreads();
    if (k0 + 32 < K) {                 // issue next slice while consuming this one
      async_ld_b128(Ag + k0 + 32, &As[buf ^ 1][srow * 32 + scol]);
      async_ld_b128(Bg + k0 + 32, &Bs[buf ^ 1][srow * 32 + scol]);
    }
    const __bf16* arow  = &As[buf][(mw * 16 + lm) * 32];
    const __bf16* brow0 = &Bs[buf][(nw * 32 + lm) * 32];
    const __bf16* brow1 = &Bs[buf][(nw * 32 + 16 + lm) * 32];

    v16bf a, b0, b1;
    v8bf t0 = ld8(arow + lh * 8);
    v8bf t1 = ld8(arow + 16 + lh * 8);
#pragma unroll
    for (int i = 0; i < 8; i++) { a[i] = t0[i]; a[8 + i] = t1[i]; }
    v8bf u0 = ld8(brow0 + lh * 8), u1 = ld8(brow0 + 16 + lh * 8);
#pragma unroll
    for (int i = 0; i < 8; i++) { b0[i] = u0[i]; b0[8 + i] = u1[i]; }
    v8bf w0 = ld8(brow1 + lh * 8), w1 = ld8(brow1 + 16 + lh * 8);
#pragma unroll
    for (int i = 0; i < 8; i++) { b1[i] = w0[i]; b1[8 + i] = w1[i]; }

    acc0 = __builtin_amdgcn_wmma_f32_16x16x32_bf16(false, a, false, b0, (short)0, acc0, false, false);
    acc1 = __builtin_amdgcn_wmma_f32_16x16x32_bf16(false, a, false, b1, (short)0, acc1, false, false);
  }
#else
  // fallback: direct global-load fragments
  const __bf16* arow  = A + (size_t)(m0 + lm) * K;
  const __bf16* brow0 = W + (size_t)(n0 + lm) * K;
  const __bf16* brow1 = W + (size_t)(n0 + 16 + lm) * K;
  for (int k0 = 0; k0 < K; k0 += 32) {
    if (k0 + 32 < K) {
      __builtin_prefetch(brow0 + k0 + 32, 0, 3);
      __builtin_prefetch(brow1 + k0 + 32, 0, 3);
    }
    v16bf a, b0, b1;
    v8bf t0 = ld8(arow + k0 + lh * 8);
    v8bf t1 = ld8(arow + k0 + 16 + lh * 8);
#pragma unroll
    for (int i = 0; i < 8; i++) { a[i] = t0[i]; a[8 + i] = t1[i]; }
    v8bf u0 = ld8(brow0 + k0 + lh * 8), u1 = ld8(brow0 + k0 + 16 + lh * 8);
#pragma unroll
    for (int i = 0; i < 8; i++) { b0[i] = u0[i]; b0[8 + i] = u1[i]; }
    v8bf w0 = ld8(brow1 + k0 + lh * 8), w1 = ld8(brow1 + k0 + 16 + lh * 8);
#pragma unroll
    for (int i = 0; i < 8; i++) { b1[i] = w0[i]; b1[8 + i] = w1[i]; }
    acc0 = __builtin_amdgcn_wmma_f32_16x16x32_bf16(false, a, false, b0, (short)0, acc0, false, false);
    acc1 = __builtin_amdgcn_wmma_f32_16x16x32_bf16(false, a, false, b1, (short)0, acc1, false, false);
  }
#endif

  if (mode == 0) {
    __bf16* out = (__bf16*)outp;
#pragma unroll
    for (int r = 0; r < 8; r++) {
      int m = m0 + lh * 8 + r;
      out[(size_t)m * N + n0 + lm]      = (__bf16)(acc0[r] * scale);
      out[(size_t)m * N + n0 + 16 + lm] = (__bf16)(acc1[r] * scale);
    }
  } else if (mode == 1) {
    __bf16* out = (__bf16*)outp;  // [N][M]; per-lane D rows are contiguous -> b128 stores
    v8bf o0, o1;
#pragma unroll
    for (int r = 0; r < 8; r++) { o0[r] = (__bf16)(acc0[r] * scale); o1[r] = (__bf16)(acc1[r] * scale); }
    *(v8bf*)(out + (size_t)(n0 + lm) * M + m0 + lh * 8)      = o0;
    *(v8bf*)(out + (size_t)(n0 + 16 + lm) * M + m0 + lh * 8) = o1;
  } else {
    float* out = (float*)outp;
#pragma unroll
    for (int r = 0; r < 8; r++) {
      int m = m0 + lh * 8 + r;
      out[(size_t)m * N + n0 + lm]      = acc0[r] * scale;
      out[(size_t)m * N + n0 + 16 + lm] = acc1[r] * scale;
    }
  }
}

// ---------------------------------------------------------------------------
// Attention: one block per 16 query rows per (b,h).
// LDS: S fp32 [16][2048] (128KB) + P bf16 [16][2048] (64KB) + scratch.
// Phase 1: S = (Q*0.125) @ K^T via WMMA, causal mask, skip fully-masked tiles.
// Phase 2: softmax in LDS; write fp32 weights to d_out; bf16 P to LDS.
// Phase 3: O = P @ V via WMMA reading A-frags from LDS, B-frags from Vt.
// ---------------------------------------------------------------------------
#define SMEM_S_OFF   0
#define SMEM_P_OFF   (16 * T_SZ * 4)                 // 131072
#define SMEM_RED_OFF (SMEM_P_OFF + 16 * T_SZ * 2)    // 196608
#define SMEM_ACC_OFF (SMEM_RED_OFF + 256 * 4)        // 197632
#define SMEM_TOTAL   (SMEM_ACC_OFF + 4 * 256 * 4)    // 201728 bytes (< 320KB WGP LDS)

__global__ __launch_bounds__(256)
void attn_softmax_wmma(const __bf16* __restrict__ Qb, const __bf16* __restrict__ Kb,
                       const __bf16* __restrict__ Vt, __bf16* __restrict__ AO,
                       float* __restrict__ Wout) {
  extern __shared__ char smem[];
  float*  S    = (float*)(smem + SMEM_S_OFF);
  __bf16* P    = (__bf16*)(smem + SMEM_P_OFF);
  float*  red  = (float*)(smem + SMEM_RED_OFF);
  float*  accb = (float*)(smem + SMEM_ACC_OFF);

  const int tid = threadIdx.x;
  const int wave = tid >> 5, lane = tid & 31, lm = lane & 15, lh = lane >> 4;
  const int q0 = blockIdx.x * 16;
  const int h  = blockIdx.y;
  const int b  = blockIdx.z;
  const float NEG = -__builtin_huge_valf();

  // ---- Phase 1: scores into LDS ----
  v16bf aq0, aq1;
  {
    const __bf16* qrow = Qb + (size_t)(b * T_SZ + q0 + lm) * C_SZ + h * DH;
    v8bf t0 = ld8(qrow + lh * 8),      t1 = ld8(qrow + 16 + lh * 8);
    v8bf t2 = ld8(qrow + 32 + lh * 8), t3 = ld8(qrow + 48 + lh * 8);
#pragma unroll
    for (int i = 0; i < 8; i++) { aq0[i] = t0[i]; aq0[8 + i] = t1[i]; aq1[i] = t2[i]; aq1[8 + i] = t3[i]; }
  }
  for (int j0 = wave * 16; j0 < T_SZ; j0 += 128) {
    if (j0 > q0 + 15) {                      // tile fully past the diagonal
#pragma unroll
      for (int r = 0; r < 8; r++) S[(r + 8 * lh) * T_SZ + j0 + lm] = NEG;
      continue;
    }
    v8f acc = {};
    const __bf16* krow = Kb + (size_t)(b * T_SZ + j0 + lm) * C_SZ + h * DH;
    {
      v16bf bk;
      v8bf u0 = ld8(krow + lh * 8), u1 = ld8(krow + 16 + lh * 8);
#pragma unroll
      for (int i = 0; i < 8; i++) { bk[i] = u0[i]; bk[8 + i] = u1[i]; }
      acc = __builtin_amdgcn_wmma_f32_16x16x32_bf16(false, aq0, false, bk, (short)0, acc, false, false);
      u0 = ld8(krow + 32 + lh * 8); u1 = ld8(krow + 48 + lh * 8);
#pragma unroll
      for (int i = 0; i < 8; i++) { bk[i] = u0[i]; bk[8 + i] = u1[i]; }
      acc = __builtin_amdgcn_wmma_f32_16x16x32_bf16(false, aq1, false, bk, (short)0, acc, false, false);
    }
#pragma unroll
    for (int r = 0; r < 8; r++) {
      int m = r + 8 * lh;
      float v = acc[r];
      if (j0 + lm > q0 + m) v = NEG;         // causal mask
      S[m * T_SZ + j0 + lm] = v;
    }
  }
  __syncthreads();

  // ---- Phase 2: softmax over each of the 16 rows (16 threads per row) ----
  {
    const int r = tid >> 4, sub = tid & 15;
    float* Srow = S + r * T_SZ;
    const int c0 = sub * 128;
    float mx = NEG;
    for (int i = 0; i < 128; i++) mx = fmaxf(mx, Srow[c0 + i]);
    red[tid] = mx; __syncthreads();
    if (sub == 0) {
      float m2 = red[r * 16];
      for (int i = 1; i < 16; i++) m2 = fmaxf(m2, red[r * 16 + i]);
      red[r * 16] = m2;
    }
    __syncthreads();
    mx = red[r * 16];
    __syncthreads();
    float lsum = 0.f;
    for (int i = 0; i < 128; i++) {
      float p = __expf(Srow[c0 + i] - mx);
      Srow[c0 + i] = p; lsum += p;
    }
    red[tid] = lsum; __syncthreads();
    if (sub == 0) {
      float s2 = 0.f;
      for (int i = 0; i < 16; i++) s2 += red[r * 16 + i];
      red[r * 16] = s2;
    }
    __syncthreads();
    const float inv = 1.0f / red[r * 16];
    float*  wrow = Wout + ((size_t)((b * H_SZ + h) * T_SZ + q0 + r)) * T_SZ;
    __bf16* prow = P + r * T_SZ;
    for (int i = 0; i < 32; i++) {
      int c = c0 + i * 4;
      float4 v = *(float4*)(Srow + c);
      v.x *= inv; v.y *= inv; v.z *= inv; v.w *= inv;
      *(float4*)(wrow + c) = v;              // fp32 softmax-weights output
      v4bf pb;
      pb[0] = (__bf16)v.x; pb[1] = (__bf16)v.y; pb[2] = (__bf16)v.z; pb[3] = (__bf16)v.w;
      *(v4bf*)(prow + c) = pb;               // bf16 P for the PV WMMA
    }
  }
  __syncthreads();

  // ---- Phase 3: O = P @ V. 4 d-tiles x 2 k-halves across the 8 waves ----
  {
    const int dtile = wave & 3, khalf = wave >> 2;
    const int kmax = min(T_SZ, (q0 + 16 + 31) & ~31);   // skip zero tail past diagonal
    const int kbeg = khalf * 1024;
    const int kend = min(kmax, (khalf + 1) * 1024);
    v8f acc = {};
    const __bf16* vrow = Vt + (size_t)(h * DH + dtile * 16 + lm) * M_SZ + b * T_SZ;
    const __bf16* prow = P + lm * T_SZ;
    for (int k0 = kbeg; k0 < kend; k0 += 32) {
      v16bf ap, bv;
      v8bf t0 = *(const v8bf*)(prow + k0 + lh * 8);      // ds_read_b128
      v8bf t1 = *(const v8bf*)(prow + k0 + 16 + lh * 8);
#pragma unroll
      for (int i = 0; i < 8; i++) { ap[i] = t0[i]; ap[8 + i] = t1[i]; }
      v8bf u0 = ld8(vrow + k0 + lh * 8), u1 = ld8(vrow + k0 + 16 + lh * 8);
#pragma unroll
      for (int i = 0; i < 8; i++) { bv[i] = u0[i]; bv[8 + i] = u1[i]; }
      acc = __builtin_amdgcn_wmma_f32_16x16x32_bf16(false, ap, false, bv, (short)0, acc, false, false);
    }
    if (khalf == 1) {
#pragma unroll
      for (int r = 0; r < 8; r++) accb[dtile * 256 + lane * 8 + r] = acc[r];
    }
    __syncthreads();
    if (khalf == 0) {
#pragma unroll
      for (int r = 0; r < 8; r++) {
        float v = acc[r] + accb[dtile * 256 + lane * 8 + r];
        int m = r + 8 * lh;
        AO[(size_t)(b * T_SZ + q0 + m) * C_SZ + h * DH + dtile * 16 + lm] = (__bf16)v;
      }
    }
  }
}

// ---------------------------------------------------------------------------
// Host orchestration
// ---------------------------------------------------------------------------
extern "C" void kernel_launch(void* const* d_in, const int* in_sizes, int n_in,
                              void* d_out, int out_size, void* d_ws, size_t ws_size,
                              hipStream_t stream) {
  (void)in_sizes; (void)n_in; (void)out_size; (void)ws_size;
  const float* x  = (const float*)d_in[0];
  const float* wq = (const float*)d_in[1];
  const float* wk = (const float*)d_in[2];
  const float* wv = (const float*)d_in[3];
  const float* wo = (const float*)d_in[4];

  const size_t XE = (size_t)M_SZ * C_SZ;   // 3,145,728 elems
  const size_t WE = (size_t)C_SZ * C_SZ;   //   589,824 elems

  __bf16* xb   = (__bf16*)d_ws;
  __bf16* wqb  = xb + XE;
  __bf16* wkb  = wqb + WE;
  __bf16* wvb  = wkb + WE;
  __bf16* wob  = wvb + WE;
  __bf16* Qb   = wob + WE;
  __bf16* Kbuf = Qb + XE;
  __bf16* Vtb  = Kbuf + XE;      // stored transposed: [C_SZ][M_SZ]
  __bf16* AO   = Vtb + XE;

  float* final_out = (float*)d_out;
  float* w_out     = final_out + XE;       // softmax weights [B,H,T,T]

  // conversions
  cvt_f32_bf16<<<dim3((unsigned)((XE / 4 + 255) / 256)), 256, 0, stream>>>(x,  xb,  (int)(XE / 4));
  cvt_f32_bf16<<<dim3((unsigned)((WE / 4 + 255) / 256)), 256, 0, stream>>>(wq, wqb, (int)(WE / 4));
  cvt_f32_bf16<<<dim3((unsigned)((WE / 4 + 255) / 256)), 256, 0, stream>>>(wk, wkb, (int)(WE / 4));
  cvt_f32_bf16<<<dim3((unsigned)((WE / 4 + 255) / 256)), 256, 0, stream>>>(wv, wvb, (int)(WE / 4));
  cvt_f32_bf16<<<dim3((unsigned)((WE / 4 + 255) / 256)), 256, 0, stream>>>(wo, wob, (int)(WE / 4));

  // Q/K/V projections (1/sqrt(Dh)=0.125 folded into Q; V stored transposed)
  dim3 gblk(M_SZ / 64, C_SZ / 64);
  gemm_xwT_wmma<<<gblk, 256, 0, stream>>>(xb, wqb, (void*)Qb,   M_SZ, C_SZ, C_SZ, 0, 0.125f);
  gemm_xwT_wmma<<<gblk, 256, 0, stream>>>(xb, wkb, (void*)Kbuf, M_SZ, C_SZ, C_SZ, 0, 1.0f);
  gemm_xwT_wmma<<<gblk, 256, 0, stream>>>(xb, wvb, (void*)Vtb,  M_SZ, C_SZ, C_SZ, 1, 1.0f);

  // attention (needs >64KB dynamic LDS; WGP has 320KB)
  (void)hipFuncSetAttribute((const void*)attn_softmax_wmma,
                            hipFuncAttributeMaxDynamicSharedMemorySize, SMEM_TOTAL);
  attn_softmax_wmma<<<dim3(T_SZ / 16, H_SZ, B_SZ), 256, SMEM_TOTAL, stream>>>(
      Qb, Kbuf, Vtb, AO, w_out);

  // final projection -> fp32 d_out
  gemm_xwT_wmma<<<gblk, 256, 0, stream>>>(AO, wob, (void*)final_out, M_SZ, C_SZ, C_SZ, 2, 1.0f);
}
